// FlashAttention_72000831750650
// MI455X (gfx1250) — compile-verified, matrix operand order checked by reading, not run on hardware
//
#include <hip/hip_runtime.h>

typedef __attribute__((ext_vector_type(16))) _Float16 v16h;
typedef __attribute__((ext_vector_type(8)))  _Float16 v8h;
typedef __attribute__((ext_vector_type(2)))  _Float16 v2h;
typedef __attribute__((ext_vector_type(8)))  float    v8f;
typedef __attribute__((ext_vector_type(4)))  float    v4f;

#define SEQ   2048
#define DHEAD 64
#define KTILE 32
#define NWAVE 4
#define MWAVE 2                      // 16-row M tiles per wave
#define QTILE (NWAVE * MWAVE * 16)   // 128 q rows per block

// ---- DPP16 16-lane butterfly reductions (stay within each 16-lane half) ----
template <int CTRL>
__device__ __forceinline__ float dpp_movf(float x) {
  int r = __builtin_amdgcn_update_dpp(0, __float_as_int(x), CTRL, 0xf, 0xf, true);
  return __int_as_float(r);
}
__device__ __forceinline__ float red_max16(float v) {
  v = fmaxf(v, dpp_movf<0xB1>(v));    // quad_perm [1,0,3,2]  (xor1)
  v = fmaxf(v, dpp_movf<0x4E>(v));    // quad_perm [2,3,0,1]  (xor2)
  v = fmaxf(v, dpp_movf<0x141>(v));   // row_half_mirror      (xor7)
  v = fmaxf(v, dpp_movf<0x140>(v));   // row_mirror           (xor15)
  return v;
}
__device__ __forceinline__ float red_sum16(float v) {
  v += dpp_movf<0xB1>(v);
  v += dpp_movf<0x4E>(v);
  v += dpp_movf<0x141>(v);
  v += dpp_movf<0x140>(v);
  return v;
}

// staged tile registers: one key pair x 8 d-values of K and V
struct StageRegs { v4f ka0, ka1, kb0, kb1, va0, va1, vb0, vb1; };

__device__ __forceinline__ void stage_load(StageRegs& sr,
                                           const float* __restrict__ kp,
                                           const float* __restrict__ vp,
                                           int kbase, int kpr, int dg8) {
  const float* k0 = kp + (size_t)(kbase + kpr) * DHEAD + dg8;
  const float* v0 = vp + (size_t)(kbase + kpr) * DHEAD + dg8;
  sr.ka0 = *(const v4f*)(k0);
  sr.ka1 = *(const v4f*)(k0 + 4);
  sr.kb0 = *(const v4f*)(k0 + DHEAD);
  sr.kb1 = *(const v4f*)(k0 + DHEAD + 4);
  sr.va0 = *(const v4f*)(v0);
  sr.va1 = *(const v4f*)(v0 + 4);
  sr.vb0 = *(const v4f*)(v0 + DHEAD);
  sr.vb1 = *(const v4f*)(v0 + DHEAD + 4);
}

__device__ __forceinline__ void stage_store(const StageRegs& sr,
                                            _Float16 (*k_l)[DHEAD],
                                            _Float16 (*vt_l)[KTILE],
                                            int kpr, int dg8) {
  v8h kh0, kh1;
  #pragma unroll
  for (int i = 0; i < 4; ++i) {
    kh0[i]     = (_Float16)sr.ka0[i];
    kh0[4 + i] = (_Float16)sr.ka1[i];
    kh1[i]     = (_Float16)sr.kb0[i];
    kh1[4 + i] = (_Float16)sr.kb1[i];
  }
  *(v8h*)&k_l[kpr][dg8]     = kh0;   // ds_store_b128
  *(v8h*)&k_l[kpr + 1][dg8] = kh1;   // ds_store_b128
  #pragma unroll
  for (int i = 0; i < 8; ++i) {
    v2h pr;
    pr[0] = (_Float16)(i < 4 ? sr.va0[i] : sr.va1[i - 4]);
    pr[1] = (_Float16)(i < 4 ? sr.vb0[i] : sr.vb1[i - 4]);
    *(v2h*)&vt_l[dg8 + i][kpr] = pr; // ds_store_b32
  }
}

__global__ __launch_bounds__(128, 1)
void fa_fwd_f16wmma(const float* __restrict__ qg,
                    const float* __restrict__ kg,
                    const float* __restrict__ vg,
                    float* __restrict__ og)
{
  __shared__ alignas(16) _Float16 k_lds[2][KTILE][DHEAD];            // double-buffered
  __shared__ alignas(16) _Float16 vt_lds[2][DHEAD][KTILE];           // double-buffered
  __shared__ alignas(16) _Float16 p_lds[NWAVE][MWAVE][16][KTILE];    // P transpose bounce

  const int tid  = threadIdx.x;
  const int wave = tid >> 5;
  const int lane = tid & 31;
  const int half = lane >> 4;
  const int l16  = lane & 15;

  const int bh    = blockIdx.x;        // 0..63 (B*H)
  const int qblk  = blockIdx.y;        // 0..15 (S/QTILE)
  const int qbase = qblk * QTILE;

  const float* qp = qg + (size_t)bh * SEQ * DHEAD;
  const float* kp = kg + (size_t)bh * SEQ * DHEAD;
  const float* vp = vg + (size_t)bh * SEQ * DHEAD;
  float*       op = og + (size_t)bh * SEQ * DHEAD;

  const int kpr = (tid & 15) * 2;      // staging: key pair base
  const int dg8 = (tid >> 4) * 8;      // staging: d group base

  // 1/sqrt(64) * log2(e): scores in base-2 domain -> native v_exp_f32 via exp2f
  const float QSCALE = 0.125f * 1.44269504088896340736f;

  // ---- Q: two M tiles x two 16x32 A-fragments, pre-scaled ----
  v16h qa[MWAVE][2];
  #pragma unroll
  for (int u = 0; u < MWAVE; ++u) {
    const float* qr = qp + (size_t)(qbase + wave * 32 + u * 16 + l16) * DHEAD;
    #pragma unroll
    for (int f = 0; f < 2; ++f) {
      const int d0 = f * 32 + 8 * half;
      const int d1 = f * 32 + 16 + 8 * half;
      #pragma unroll
      for (int i = 0; i < 8; ++i) {
        qa[u][f][i]     = (_Float16)(qr[d0 + i] * QSCALE);
        qa[u][f][8 + i] = (_Float16)(qr[d1 + i] * QSCALE);
      }
    }
  }

  v8f o_acc[MWAVE][4] = {{{}, {}, {}, {}}, {{}, {}, {}, {}}};
  float m_i[MWAVE][8], l_i[MWAVE][8];
  #pragma unroll
  for (int u = 0; u < MWAVE; ++u)
    #pragma unroll
    for (int r = 0; r < 8; ++r) { m_i[u][r] = -1e30f; l_i[u][r] = 0.0f; }

  const int kb_end    = (qbase + QTILE) / KTILE;            // block causal limit
  const int my_kb_end = qbase / KTILE + wave + 1;           // this wave's causal limit

  // ---- pipeline prologue: stage tile 0 ----
  StageRegs sr;
  stage_load(sr, kp, vp, 0, kpr, dg8);
  stage_store(sr, k_lds[0], vt_lds[0], kpr, dg8);
  __syncthreads();

  for (int kb = 0; kb < kb_end; ++kb) {
    const int kbase = kb * KTILE;
    const int buf   = kb & 1;

    // issue next tile's global loads before compute (overlaps with WMMAs)
    if (kb + 1 < kb_end) stage_load(sr, kp, vp, kbase + KTILE, kpr, dg8);
    if (kb + 2 < kb_end) {
      __builtin_prefetch(kp + (size_t)(kbase + 2 * KTILE + kpr) * DHEAD + dg8, 0, 1);
      __builtin_prefetch(vp + (size_t)(kbase + 2 * KTILE + kpr) * DHEAD + dg8, 0, 1);
    }

    if (kb < my_kb_end) {   // wave-uniform: skip fully-masked tail blocks
      // ---- S = Q K^T : each B fragment feeds both M tiles ----
      v8f s[MWAVE][2] = {{{}, {}}, {{}, {}}};
      #pragma unroll
      for (int t = 0; t < 2; ++t) {
        #pragma unroll
        for (int f = 0; f < 2; ++f) {
          v8h lo = *(const v8h*)&k_lds[buf][16 * t + l16][f * 32 + 8 * half];
          v8h hi = *(const v8h*)&k_lds[buf][16 * t + l16][f * 32 + 16 + 8 * half];
          v16h b;
          #pragma unroll
          for (int i = 0; i < 8; ++i) { b[i] = lo[i]; b[8 + i] = hi[i]; }
          #pragma unroll
          for (int u = 0; u < MWAVE; ++u)
            s[u][t] = __builtin_amdgcn_wmma_f32_16x16x32_f16(false, qa[u][f], false, b,
                                                             (short)0, s[u][t], false, false);
        }
      }

      // ---- mask (only near the causal diagonal) + online softmax ----
      float sc_r[MWAVE][8];
      #pragma unroll
      for (int u = 0; u < MWAVE; ++u) {
        const int tbase = qbase + wave * 32 + u * 16;   // min row of this tile
        if (kbase + KTILE - 1 > tbase) {                // wave-uniform branch
          const int mrow = tbase + 8 * half;
          #pragma unroll
          for (int t = 0; t < 2; ++t) {
            const int ncol = kbase + 16 * t + l16;
            #pragma unroll
            for (int r = 0; r < 8; ++r)
              if (ncol > mrow + r) s[u][t][r] = -1e30f;
          }
        }
        #pragma unroll
        for (int r = 0; r < 8; ++r) {
          const float mx   = red_max16(fmaxf(s[u][0][r], s[u][1][r]));
          const float mnew = fmaxf(m_i[u][r], mx);
          const float sc   = exp2f(m_i[u][r] - mnew);
          const float p0   = exp2f(s[u][0][r] - mnew);
          const float p1   = exp2f(s[u][1][r] - mnew);
          s[u][0][r] = p0;
          s[u][1][r] = p1;
          const float rs = red_sum16(p0 + p1);
          l_i[u][r]  = l_i[u][r] * sc + rs;
          m_i[u][r]  = mnew;
          sc_r[u][r] = sc;
        }
      }

      // ---- C-layout P -> A-layout P via per-tile LDS bounce ----
      v16h pa[MWAVE];
      #pragma unroll
      for (int u = 0; u < MWAVE; ++u) {
        #pragma unroll
        for (int t = 0; t < 2; ++t)
          #pragma unroll
          for (int r = 0; r < 8; ++r)
            p_lds[wave][u][8 * half + r][16 * t + l16] = (_Float16)s[u][t][r];
        v8h lo = *(const v8h*)&p_lds[wave][u][l16][8 * half];
        v8h hi = *(const v8h*)&p_lds[wave][u][l16][16 + 8 * half];
        #pragma unroll
        for (int i = 0; i < 8; ++i) { pa[u][i] = lo[i]; pa[u][8 + i] = hi[i]; }
      }

      // ---- O = O*scale + P @ V : each V fragment feeds both M tiles ----
      #pragma unroll
      for (int c = 0; c < 4; ++c) {
        v8h lo = *(const v8h*)&vt_lds[buf][16 * c + l16][8 * half];
        v8h hi = *(const v8h*)&vt_lds[buf][16 * c + l16][16 + 8 * half];
        v16h bv;
        #pragma unroll
        for (int i = 0; i < 8; ++i) { bv[i] = lo[i]; bv[8 + i] = hi[i]; }
        #pragma unroll
        for (int u = 0; u < MWAVE; ++u) {
          #pragma unroll
          for (int r = 0; r < 8; ++r) o_acc[u][c][r] *= sc_r[u][r];
          o_acc[u][c] = __builtin_amdgcn_wmma_f32_16x16x32_f16(false, pa[u], false, bv,
                                                               (short)0, o_acc[u][c], false, false);
        }
      }
    }

    // convert + store next tile into the alternate buffer
    if (kb + 1 < kb_end)
      stage_store(sr, k_lds[buf ^ 1], vt_lds[buf ^ 1], kpr, dg8);

    __syncthreads();   // single barrier per iteration
  }

  // ---- epilogue: normalize and store ----
  #pragma unroll
  for (int u = 0; u < MWAVE; ++u) {
    #pragma unroll
    for (int r = 0; r < 8; ++r) l_i[u][r] = 1.0f / l_i[u][r];
    const int orow = qbase + wave * 32 + u * 16 + 8 * half;
    #pragma unroll
    for (int c = 0; c < 4; ++c)
      #pragma unroll
      for (int r = 0; r < 8; ++r)
        op[(size_t)(orow + r) * DHEAD + 16 * c + l16] = o_acc[u][c][r] * l_i[u][r];
  }
}

extern "C" void kernel_launch(void* const* d_in, const int* in_sizes, int n_in,
                              void* d_out, int out_size, void* d_ws, size_t ws_size,
                              hipStream_t stream) {
  (void)in_sizes; (void)n_in; (void)out_size; (void)d_ws; (void)ws_size;
  const float* q = (const float*)d_in[0];
  const float* k = (const float*)d_in[1];
  const float* v = (const float*)d_in[2];
  float* out = (float*)d_out;
  dim3 grid(64 /* B*H */, SEQ / QTILE);
  fa_fwd_f16wmma<<<grid, NWAVE * 32, 0, stream>>>(q, k, v, out);
}
